// CACMN_83150566851085
// MI455X (gfx1250) — matrix operand order repaired and checked by reading, not compiled, not run on hardware
//
#include <hip/hip_runtime.h>
#include <stdint.h>
#include <stddef.h>

// ---------------------------------------------------------------------------
// CDNA5 (gfx1250) implementation: bf16 WMMA GEMMs + fused GRU/attention ops.
// Shapes: B=64, S=100, QMAX=10, E=256, H=256.  3H = 768.
// GEMM: 8 waves/block, block tile 32x256, wave tile 16x64 (4 WMMA accumulators
// sharing one A fragment per K-step for 1.6x higher arithmetic intensity).
// ---------------------------------------------------------------------------

typedef __attribute__((ext_vector_type(16))) __bf16 v16bf;
typedef __attribute__((ext_vector_type(8)))  float  v8f;

#define BB   64
#define SS   100
#define QMAX 10
#define EE   256
#define HH   256
#define H3   768
#define BS   (BB * SS)          // 6400

__device__ __forceinline__ uint16_t f32_to_bf16(float f) {
    union { float f; uint32_t u; } c; c.f = f;
    uint32_t u = c.u;
    return (uint16_t)((u + 0x7FFFu + ((u >> 16) & 1u)) >> 16);
}

__device__ __forceinline__ float sigmoidf(float x) {
    return 1.0f / (1.0f + __expf(-x));
}

// Block-wide sum (blockDim.x == 256, 8 waves of 32). Returns result to all threads.
__device__ __forceinline__ float blockReduceSum(float v, float* red) {
    __syncthreads();                       // protect red[] reuse across calls
    #pragma unroll
    for (int o = 16; o > 0; o >>= 1) v += __shfl_down(v, o, 32);
    const int lane = threadIdx.x & 31;
    const int w    = threadIdx.x >> 5;
    if (lane == 0) red[w] = v;
    __syncthreads();
    float s = 0.0f;
    if (w == 0) {
        s = (lane < 8) ? red[lane] : 0.0f;
        #pragma unroll
        for (int o = 4; o > 0; o >>= 1) s += __shfl_down(s, o, 32);
        if (lane == 0) red[0] = s;
    }
    __syncthreads();
    return red[0];
}

// ---------------------------------------------------------------------------
// GEMM: C[M,N] = act( A[M,K](bf16) * Bt[N,K]^T(bf16) + bias[N] )
// Block tile 32x256 (grid: x over N/256, y over M/32); wave tile 16x64.
// Fragment layouts per CDNA5 ISA 7.12.2 (16-bit A 16x32; B 32x16).
// ---------------------------------------------------------------------------
__global__ __launch_bounds__(256) void gemm_bf16_wmma(
    const uint16_t* __restrict__ A, const uint16_t* __restrict__ Bt,
    const float* __restrict__ bias, float* __restrict__ C,
    int M, int N, int K, int act)
{
    const int wave = threadIdx.x >> 5;
    const int lane = threadIdx.x & 31;
    const int m0 = blockIdx.y * 32 + (wave >> 2) * 16;   // 2 M-tiles per block
    const int n0 = blockIdx.x * 256 + (wave & 3) * 64;   // 4 N-groups per block
    if (m0 >= M || n0 >= N) return;        // uniform per wave -> EXEC all-1s in WMMA

    const int half = lane >> 4;            // lane half selects K sub-range
    const int r16  = lane & 15;

    // A: lane half 0 -> K pairs {0..7, 16..23}; half 1 -> {8..15, 24..31}
    const uint16_t* Arow = A  + (size_t)(m0 + r16) * K + half * 8;
    // B: lane half 0 -> K pairs {0..7, 16..23}; half 1 -> {8..15, 24..31}
    const uint16_t* Brow = Bt + (size_t)(n0 + r16) * K + half * 8;
    const size_t bstride = (size_t)16 * K;               // next 16-column group

    v8f acc0 = {}, acc1 = {}, acc2 = {}, acc3 = {};
    for (int k = 0; k < K; k += 32) {
        union { uint32_t u[8]; v16bf v; } af, b0, b1, b2, b3;
        #pragma unroll
        for (int v = 0; v < 8; ++v) {
            const int kpa = (v < 4) ? (2 * v) : (2 * v + 8);
            af.u[v] = *(const uint32_t*)(Arow + k + kpa);
            b0.u[v] = *(const uint32_t*)(Brow + 0 * bstride + k + kpa);
            b1.u[v] = *(const uint32_t*)(Brow + 1 * bstride + k + kpa);
            b2.u[v] = *(const uint32_t*)(Brow + 2 * bstride + k + kpa);
            b3.u[v] = *(const uint32_t*)(Brow + 3 * bstride + k + kpa);
        }
        if (k + 32 < K) {                  // speculative prefetch of next K tile
            __builtin_prefetch(Arow + k + 32, 0, 1);
            __builtin_prefetch(Brow + k + 32, 0, 1);
        }
        acc0 = __builtin_amdgcn_wmma_f32_16x16x32_bf16(false, af.v, false, b0.v, (short)0, acc0, false, false);
        acc1 = __builtin_amdgcn_wmma_f32_16x16x32_bf16(false, af.v, false, b1.v, (short)0, acc1, false, false);
        acc2 = __builtin_amdgcn_wmma_f32_16x16x32_bf16(false, af.v, false, b2.v, (short)0, acc2, false, false);
        acc3 = __builtin_amdgcn_wmma_f32_16x16x32_bf16(false, af.v, false, b3.v, (short)0, acc3, false, false);
    }

    // C/D layout: VGPR r, lanes 0-15 -> M=r, lanes 16-31 -> M=r+8; N = lane&15
    v8f accs[4] = {acc0, acc1, acc2, acc3};
    #pragma unroll
    for (int g = 0; g < 4; ++g) {
        const int col = n0 + g * 16 + r16;
        #pragma unroll
        for (int r = 0; r < 8; ++r) {
            const int row = m0 + r + half * 8;
            float v = accs[g][r] + (bias ? bias[col] : 0.0f);
            if (act == 1) v = tanhf(v);
            C[(size_t)row * N + col] = v;
        }
    }
}

// ---------------------------------------------------------------------------
// GRU gate fusion: h' = (1-z)*n + z*h ; writes f32 h' and bf16 copy for next GEMM
// gx/gh: [M, 768] = [z | r | n] thirds. hprev/hout/hbf: [M, 256]
// ---------------------------------------------------------------------------
__global__ void gru_gate(const float* __restrict__ gx, const float* __restrict__ gh,
                         const float* __restrict__ hprev, float* __restrict__ hout,
                         uint16_t* __restrict__ hbf, int M)
{
    const int tid = blockIdx.x * 256 + threadIdx.x;
    if (tid >= M * HH) return;
    const int m = tid >> 8;
    const int h = tid & 255;
    const size_t g = (size_t)m * H3 + h;
    const float z = sigmoidf(gx[g]         + gh[g]);
    const float r = sigmoidf(gx[g + HH]    + gh[g + HH]);
    const float n = tanhf   (gx[g + 2*HH]  + r * gh[g + 2*HH]);
    const float hn = (1.0f - z) * n + z * hprev[tid];
    hout[tid] = hn;
    hbf[tid]  = f32_to_bf16(hn);
}

// ---------------------------------------------------------------------------
// Gathers (embedding lookups -> bf16 GEMM operands)
// ---------------------------------------------------------------------------
__global__ void gather_k(const int* __restrict__ kv, const float* __restrict__ Eq,
                         uint16_t* __restrict__ kX)   // [QMAX, BS, E] time-major
{
    const int tid = blockIdx.x * 256 + threadIdx.x;
    if (tid >= QMAX * BS * EE) return;
    const int e  = tid & 255;
    const int bs = (tid >> 8) % BS;
    const int q  = tid / (BS * EE);
    const int id = kv[bs * QMAX + q];
    kX[tid] = f32_to_bf16(Eq[(size_t)id * EE + e]);
}

__global__ void gather_s(const int* __restrict__ iv, const float* __restrict__ Eq,
                         const float* __restrict__ Eu, const float* __restrict__ Ev,
                         const float* __restrict__ Ec, uint16_t* __restrict__ sX) // [S, B, 4E]
{
    const int tid = blockIdx.x * 256 + threadIdx.x;
    if (tid >= SS * BB * 4 * EE) return;
    const int c = tid & 1023;
    const int b = (tid >> 10) & 63;
    const int t = tid >> 16;
    const int seg = c >> 8, e = c & 255;
    const int* row = iv + (b * SS + t) * 4;
    const float* tab; int id;
    switch (seg) {
        case 0:  tab = Eq; id = row[0]; break;
        case 1:  tab = Eu; id = row[1]; break;
        case 2:  tab = Ev; id = row[2]; break;
        default: tab = Ec; id = row[3]; break;
    }
    sX[tid] = f32_to_bf16(tab[(size_t)id * EE + e]);
}

__global__ void gather_d(const int* __restrict__ dv, const float* __restrict__ Eq,
                         const float* __restrict__ Eu, const float* __restrict__ Ev,
                         const float* __restrict__ Ec, uint16_t* __restrict__ dX) // [BS, 4E]
{
    const int tid = blockIdx.x * 256 + threadIdx.x;
    if (tid >= BS * 4 * EE) return;
    const int c  = tid & 1023;
    const int bs = tid >> 10;
    const int seg = c >> 8, e = c & 255;
    const int* row = dv + bs * 4;
    const float* tab; int id;
    switch (seg) {
        case 0:  tab = Eq; id = row[0]; break;
        case 1:  tab = Eu; id = row[1]; break;
        case 2:  tab = Ev; id = row[2]; break;
        default: tab = Ec; id = row[3]; break;
    }
    dX[tid] = f32_to_bf16(tab[(size_t)id * EE + e]);
}

__global__ void gather_e(const int* __restrict__ ev, const float* __restrict__ Ev,
                         const float* __restrict__ Ec, uint16_t* __restrict__ eX) // [10pos, 640, 3E]
{
    const int tid = blockIdx.x * 256 + threadIdx.x;
    if (tid >= QMAX * (BB * 10) * 3 * EE) return;
    const int c   = tid % H3;
    const int bc  = (tid / H3) % (BB * 10);
    const int pos = tid / (H3 * BB * 10);
    const int b = bc / 10, chunk = bc % 10;
    const int s = chunk * 10 + pos;
    const int* row = ev + (b * SS + s) * 4;
    const int seg = c / EE, e = c % EE;
    float val;
    if      (seg == 0) val = Ev[(size_t)row[2] * EE + e];
    else if (seg == 1) val = Ec[(size_t)row[3] * EE + e];
    else               val = Ec[(size_t)row[1] * EE + e];
    eX[tid] = f32_to_bf16(val);
}

// Weight transpose + cast: W[K,N] f32 -> WT[N,K] bf16
__global__ void transpose_to_bf16(const float* __restrict__ W, uint16_t* __restrict__ WT,
                                  int K, int N)
{
    const int tid = blockIdx.x * 256 + threadIdx.x;
    if (tid >= K * N) return;
    const int k = tid / N, n = tid % N;
    WT[(size_t)n * K + k] = f32_to_bf16(W[(size_t)k * N + n]);
}

// ---------------------------------------------------------------------------
// Knowledge attention: scores over QMAX, masked softmax, weighted sum.
// k_outs: [QMAX, BS, H] time-major ; kout: [BS, H]
// ---------------------------------------------------------------------------
__global__ __launch_bounds__(256) void knowledge_attn(const float* __restrict__ k_outs,
                                                      float* __restrict__ kout)
{
    __shared__ float w[QMAX];
    __shared__ float red[8];
    const int bs = blockIdx.x;
    const int s  = bs % SS;
    const int qlen = s / 10 + 1;
    const int h = threadIdx.x;
    const float hid = k_outs[(size_t)(qlen - 1) * BS * HH + (size_t)bs * HH + h];
    float ov[QMAX];
    for (int q = 0; q < QMAX; ++q) {
        ov[q] = k_outs[(size_t)q * BS * HH + (size_t)bs * HH + h];
        const float sum = blockReduceSum(ov[q] * hid, red);
        if (threadIdx.x == 0) w[q] = (q < qlen) ? sum : -1.0e9f;
    }
    __syncthreads();
    if (threadIdx.x == 0) {
        float m = -3.0e38f;
        for (int q = 0; q < QMAX; ++q) m = fmaxf(m, w[q]);
        float Z = 0.0f;
        for (int q = 0; q < QMAX; ++q) { w[q] = __expf(w[q] - m); Z += w[q]; }
        const float inv = 1.0f / Z;
        for (int q = 0; q < QMAX; ++q) w[q] *= inv;
    }
    __syncthreads();
    float acc = 0.0f;
    for (int q = 0; q < QMAX; ++q) acc += w[q] * ov[q];
    kout[(size_t)bs * HH + h] = acc;
}

// ---------------------------------------------------------------------------
// Causal self-attention over S. s_outs: [S, B, H] time-major; iout: [BS, H]
// ---------------------------------------------------------------------------
__global__ __launch_bounds__(256) void session_attn(const float* __restrict__ s_outs,
                                                    float* __restrict__ iout)
{
    __shared__ float sc[SS];
    __shared__ float red[8];
    const int bs = blockIdx.x;
    const int b = bs / SS, s = bs % SS;
    const int h = threadIdx.x;
    const float q = s_outs[((size_t)s * BB + b) * HH + h];
    for (int t = 0; t <= s; ++t) {
        const float sum = blockReduceSum(q * s_outs[((size_t)t * BB + b) * HH + h], red);
        if (threadIdx.x == 0) sc[t] = sum;
    }
    __syncthreads();
    if (threadIdx.x == 0) {
        float m = -3.0e38f;
        for (int t = 0; t <= s; ++t) m = fmaxf(m, sc[t]);
        float Z = 0.0f;
        for (int t = 0; t <= s; ++t) { sc[t] = __expf(sc[t] - m); Z += sc[t]; }
        const float inv = 1.0f / Z;
        for (int t = 0; t <= s; ++t) sc[t] *= inv;
    }
    __syncthreads();
    float acc = 0.0f;
    for (int t = 0; t <= s; ++t) acc += sc[t] * s_outs[((size_t)t * BB + b) * HH + h];
    iout[(size_t)bs * HH + h] = acc;
}

// concat [kout | iout | dout] -> bf16 [BS, 768]
__global__ void concat_to_bf16(const float* __restrict__ kout, const float* __restrict__ iout,
                               const float* __restrict__ dout, uint16_t* __restrict__ cc)
{
    const int tid = blockIdx.x * 256 + threadIdx.x;
    if (tid >= BS * H3) return;
    const int bs = tid / H3, c = tid % H3;
    const int seg = c / HH, e = c % HH;
    const float* src = (seg == 0) ? kout : (seg == 1) ? iout : dout;
    cc[tid] = f32_to_bf16(src[(size_t)bs * HH + e]);
}

// relevance = sigmoid(relh[bs,:] . rW2 + rb2)
__global__ __launch_bounds__(256) void rel_out(const float* __restrict__ A,
                                               const float* __restrict__ w,
                                               const float* __restrict__ b,
                                               float* __restrict__ out)
{
    __shared__ float red[8];
    const int m = blockIdx.x;
    const float s = blockReduceSum(A[(size_t)m * HH + threadIdx.x] * w[threadIdx.x], red);
    if (threadIdx.x == 0) out[m] = sigmoidf(s + b[0]);
}

// exam_prob with (pos, b, chunk) -> (b, chunk*10+pos) remap
__global__ __launch_bounds__(256) void exam_out(const float* __restrict__ e_outs,
                                                const float* __restrict__ w,
                                                const float* __restrict__ b,
                                                float* __restrict__ exam)
{
    __shared__ float red[8];
    const int idx = blockIdx.x;                 // [pos, bc] flattened, 6400
    const int pos = idx / (BB * 10);
    const int bc  = idx % (BB * 10);
    const int bb = bc / 10, chunk = bc % 10;
    const float s = blockReduceSum(e_outs[(size_t)idx * HH + threadIdx.x] * w[threadIdx.x], red);
    if (threadIdx.x == 0) exam[bb * SS + chunk * 10 + pos] = sigmoidf(s + b[0]);
}

__global__ void finalize_out(const float* __restrict__ rel, const float* __restrict__ exam,
                             float* __restrict__ out)
{
    const int t = blockIdx.x * 256 + threadIdx.x;
    if (t >= BS) return;
    out[t]            = rel[t];
    out[BS + t]       = exam[t];
    out[2 * BS + t]   = rel[t] * exam[t];
}

__global__ void fill_f32(float* p, int n)    { int t = blockIdx.x*256+threadIdx.x; if (t < n) p[t] = 0.0f; }
__global__ void fill_u16(uint16_t* p, int n) { int t = blockIdx.x*256+threadIdx.x; if (t < n) p[t] = 0;    }

// ---------------------------------------------------------------------------
// Host orchestration
// ---------------------------------------------------------------------------
static inline dim3 grid1(int n) { return dim3((n + 255) / 256); }

extern "C" void kernel_launch(void* const* d_in, const int* in_sizes, int n_in,
                              void* d_out, int out_size, void* d_ws, size_t ws_size,
                              hipStream_t stream) {
    (void)in_sizes; (void)n_in; (void)out_size; (void)ws_size;

    const int*   kvar = (const int*)  d_in[0];
    const int*   ivar = (const int*)  d_in[1];
    const int*   dvar = (const int*)  d_in[2];
    const int*   evar = (const int*)  d_in[3];
    const float* Eq   = (const float*)d_in[5];
    const float* Eu   = (const float*)d_in[6];
    const float* Ev   = (const float*)d_in[7];
    const float* Ec   = (const float*)d_in[8];
    const float* kWx = (const float*)d_in[9],  *kWh = (const float*)d_in[10];
    const float* kbx = (const float*)d_in[11], *kbh = (const float*)d_in[12];
    const float* sWx = (const float*)d_in[13], *sWh = (const float*)d_in[14];
    const float* sbx = (const float*)d_in[15], *sbh = (const float*)d_in[16];
    const float* dW  = (const float*)d_in[17], *db  = (const float*)d_in[18];
    const float* rW1 = (const float*)d_in[19], *rb1 = (const float*)d_in[20];
    const float* rW2 = (const float*)d_in[21], *rb2 = (const float*)d_in[22];
    const float* eWx = (const float*)d_in[23], *eWh = (const float*)d_in[24];
    const float* ebx = (const float*)d_in[25], *ebh = (const float*)d_in[26];
    const float* eWo = (const float*)d_in[27], *ebo = (const float*)d_in[28];

    char* ws = (char*)d_ws;
    size_t off = 0;
    auto alloc = [&](size_t bytes) -> char* {
        off = (off + 255) & ~(size_t)255;
        char* p = ws + off; off += bytes; return p;
    };

    // bf16 transposed weights [N,K]
    uint16_t* kWxT = (uint16_t*)alloc((size_t)H3 * EE * 2);
    uint16_t* kWhT = (uint16_t*)alloc((size_t)H3 * HH * 2);
    uint16_t* sWxT = (uint16_t*)alloc((size_t)H3 * 4 * EE * 2);
    uint16_t* sWhT = (uint16_t*)alloc((size_t)H3 * HH * 2);
    uint16_t* dWT  = (uint16_t*)alloc((size_t)HH * 4 * EE * 2);
    uint16_t* rW1T = (uint16_t*)alloc((size_t)HH * H3 * 2);
    uint16_t* eWxT = (uint16_t*)alloc((size_t)H3 * H3 * 2);
    uint16_t* eWhT = (uint16_t*)alloc((size_t)H3 * HH * 2);
    // activations / scratch
    uint16_t* kX    = (uint16_t*)alloc((size_t)QMAX * BS * EE * 2);
    float*    gxA   = (float*)   alloc((size_t)BS * H3 * 4);   // reused: k-gx / s-gx-all / e-gx-all
    float*    gxB   = (float*)   alloc((size_t)BS * H3 * 4);   // reused: per-step gh
    uint16_t* hbf   = (uint16_t*)alloc((size_t)BS * HH * 2);   // bf16 hidden (max batch 6400)
    float*    k_outs= (float*)   alloc((size_t)QMAX * BS * HH * 4);
    float*    kout  = (float*)   alloc((size_t)BS * HH * 4);
    uint16_t* sX    = (uint16_t*)alloc((size_t)SS * BB * 4 * EE * 2);
    float*    s_outs= (float*)   alloc((size_t)SS * BB * HH * 4);
    float*    iout  = (float*)   alloc((size_t)BS * HH * 4);
    uint16_t* dX    = (uint16_t*)alloc((size_t)BS * 4 * EE * 2);
    float*    dout  = (float*)   alloc((size_t)BS * HH * 4);
    uint16_t* cc    = (uint16_t*)alloc((size_t)BS * H3 * 2);
    float*    relh  = (float*)   alloc((size_t)BS * HH * 4);
    float*    relv  = (float*)   alloc((size_t)BS * 4);
    uint16_t* eX    = (uint16_t*)alloc((size_t)QMAX * BB * 10 * H3 * 2);
    float*    e_outs= (float*)   alloc((size_t)QMAX * BB * 10 * HH * 4);
    float*    examv = (float*)   alloc((size_t)BS * 4);
    float*    zeroF = (float*)   alloc((size_t)BS * HH * 4);
    uint16_t* zeroB = (uint16_t*)alloc((size_t)BS * HH * 2);

    auto gemm = [&](const uint16_t* A, const uint16_t* Bt, const float* bias,
                    float* C, int M, int N, int K, int act) {
        dim3 g((N + 255) / 256, (M + 31) / 32);
        gemm_bf16_wmma<<<g, 256, 0, stream>>>(A, Bt, bias, C, M, N, K, act);
    };

    // init zero hidden states
    fill_f32<<<grid1(BS * HH), 256, 0, stream>>>(zeroF, BS * HH);
    fill_u16<<<grid1(BS * HH), 256, 0, stream>>>(zeroB, BS * HH);

    // weight casts / transposes
    transpose_to_bf16<<<grid1(EE * H3), 256, 0, stream>>>(kWx, kWxT, EE, H3);
    transpose_to_bf16<<<grid1(HH * H3), 256, 0, stream>>>(kWh, kWhT, HH, H3);
    transpose_to_bf16<<<grid1(4 * EE * H3), 256, 0, stream>>>(sWx, sWxT, 4 * EE, H3);
    transpose_to_bf16<<<grid1(HH * H3), 256, 0, stream>>>(sWh, sWhT, HH, H3);
    transpose_to_bf16<<<grid1(4 * EE * HH), 256, 0, stream>>>(dW, dWT, 4 * EE, HH);
    transpose_to_bf16<<<grid1(H3 * HH), 256, 0, stream>>>(rW1, rW1T, H3, HH);
    transpose_to_bf16<<<grid1(H3 * H3), 256, 0, stream>>>(eWx, eWxT, H3, H3);
    transpose_to_bf16<<<grid1(HH * H3), 256, 0, stream>>>(eWh, eWhT, HH, H3);

    // ---------------- Knowledge GRU (10 steps, batch BS) ----------------
    gather_k<<<grid1(QMAX * BS * EE), 256, 0, stream>>>(kvar, Eq, kX);
    for (int q = 0; q < QMAX; ++q) {
        const uint16_t* xq = kX + (size_t)q * BS * EE;
        gemm(xq, kWxT, kbx, gxA, BS, H3, EE, 0);
        gemm(q == 0 ? zeroB : hbf, kWhT, kbh, gxB, BS, H3, HH, 0);
        const float* hp = (q == 0) ? zeroF : (k_outs + (size_t)(q - 1) * BS * HH);
        gru_gate<<<grid1(BS * HH), 256, 0, stream>>>(gxA, gxB, hp,
                                                     k_outs + (size_t)q * BS * HH, hbf, BS);
    }
    knowledge_attn<<<BS, 256, 0, stream>>>(k_outs, kout);

    // ---------------- Session GRU (100 steps, batch B) ----------------
    gather_s<<<grid1(SS * BB * 4 * EE), 256, 0, stream>>>(ivar, Eq, Eu, Ev, Ec, sX);
    gemm(sX, sWxT, sbx, gxA, BS, H3, 4 * EE, 0);       // all-step x@Wx in one GEMM
    for (int t = 0; t < SS; ++t) {
        gemm(t == 0 ? zeroB : hbf, sWhT, sbh, gxB, BB, H3, HH, 0);
        const float* hp = (t == 0) ? zeroF : (s_outs + (size_t)(t - 1) * BB * HH);
        gru_gate<<<grid1(BB * HH), 256, 0, stream>>>(gxA + (size_t)t * BB * H3, gxB, hp,
                                                     s_outs + (size_t)t * BB * HH, hbf, BB);
    }
    session_attn<<<BS, 256, 0, stream>>>(s_outs, iout);

    // ---------------- Document branch ----------------
    gather_d<<<grid1(BS * 4 * EE), 256, 0, stream>>>(dvar, Eq, Eu, Ev, Ec, dX);
    gemm(dX, dWT, db, dout, BS, HH, 4 * EE, 1);        // tanh

    // ---------------- Relevance MLP ----------------
    concat_to_bf16<<<grid1(BS * H3), 256, 0, stream>>>(kout, iout, dout, cc);
    gemm(cc, rW1T, rb1, relh, BS, HH, H3, 1);          // tanh
    rel_out<<<BS, 256, 0, stream>>>(relh, rW2, rb2, relv);

    // ---------------- Examination GRU (10 steps, batch 640) ----------------
    gather_e<<<grid1(QMAX * BB * 10 * H3), 256, 0, stream>>>(evar, Ev, Ec, eX);
    gemm(eX, eWxT, ebx, gxA, QMAX * BB * 10, H3, H3, 0);
    const int EB = BB * 10;                            // 640
    for (int p = 0; p < QMAX; ++p) {
        gemm(p == 0 ? zeroB : hbf, eWhT, ebh, gxB, EB, H3, HH, 0);
        const float* hp = (p == 0) ? zeroF : (e_outs + (size_t)(p - 1) * EB * HH);
        gru_gate<<<grid1(EB * HH), 256, 0, stream>>>(gxA + (size_t)p * EB * H3, gxB, hp,
                                                     e_outs + (size_t)p * EB * HH, hbf, EB);
    }
    exam_out<<<BS, 256, 0, stream>>>(e_outs, eWo, ebo, examv);

    // ---------------- Outputs ----------------
    finalize_out<<<grid1(BS), 256, 0, stream>>>(relv, examv, (float*)d_out);
}
